// EquivWeightQueryModel_4114578670365
// MI455X (gfx1250) — compile-verified, matrix operand order checked by reading, not run on hardware
//
#include <hip/hip_runtime.h>
#include <hip/hip_bf16.h>
#include <math.h>

typedef __attribute__((ext_vector_type(16))) _Float16 v16h;
typedef __attribute__((ext_vector_type(8)))  float    v8f;

#define WPB 8  // waves per 256-thread block (wave32)

__device__ __forceinline__ int fkey(float f) {
  int i = __float_as_int(f);
  return (i < 0) ? (i ^ 0x7fffffff) : i;
}
__device__ __forceinline__ float fdec(int k) {
  if (k < 0) k ^= 0x7fffffff;
  return __int_as_float(k);
}

// ---------------- init: zero acc/den, mx=-inf key, q = feat @ Wq ----------------
__global__ void kInit(float* __restrict__ acc, float* __restrict__ den,
                      int* __restrict__ mxk, float* __restrict__ q,
                      const float* __restrict__ feat, const float* __restrict__ Wq,
                      int N) {
  int idx = blockIdx.x * blockDim.x + threadIdx.x;
  int M = N * 86;
  if (idx < M) acc[idx] = 0.0f;
  if (idx < N) {
    den[idx] = 0.0f;
    mxk[idx] = (int)0x80000000;
    float f0 = feat[3 * idx], f1 = feat[3 * idx + 1], f2 = feat[3 * idx + 2];
#pragma unroll
    for (int u = 0; u < 10; ++u)
      q[idx * 10 + u] = f0 * Wq[u] + f1 * Wq[10 + u] + f2 * Wq[20 + u];
  }
}

// ------------- lay out Wr_k / Wr_v as fp16 WMMA B-fragments (96x32 padded) -------------
// frag index order (i fastest): [m][kt][nt][lane][i], 2*3*2*32*16 = 6144 halves
__global__ void kFrags(_Float16* __restrict__ frag,
                       const float* __restrict__ Wk, const float* __restrict__ Wv) {
  int idx = blockIdx.x * blockDim.x + threadIdx.x;
  if (idx >= 6144) return;
  int t = idx;
  int i = t & 15;  t >>= 4;
  int lane = t & 31; t >>= 5;
  int nt = t & 1;  t >>= 1;
  int kt = t % 3;  t /= 3;
  int m = t;
  int n = nt * 16 + (lane & 15);      // output channel 0..31 (28 real)
  float val = 0.0f;
  if (n < 28) {
    int kk = kt * 32 + ((lane >= 16) ? 8 : 0) + ((i >= 8) ? 16 : 0) + (i & 7); // K 0..95
    int j = kk / 3, c = kk - 3 * j;   // g index = 3*j + c
    const float* W = m ? Wv : Wk;     // [32,84] row major, col = n*3 + c
    val = W[j * 84 + n * 3 + c];
  }
  frag[idx] = (_Float16)val;
}

// ---------------- edge kernel: one wave handles a 16-edge tile ----------------
template <bool FULL>
__global__ __launch_bounds__(256) void kEdge(
    const float* __restrict__ pos, const float* __restrict__ feat,
    const float* __restrict__ Wr_h, const float* __restrict__ br_h,
    const int* __restrict__ esrc, const int* __restrict__ edst,
    const float* __restrict__ q, const _Float16* __restrict__ frag,
    float* __restrict__ acc, float* __restrict__ den, int* __restrict__ mxk,
    int N, int E) {
  __shared__ float s_wrh[320];
  __shared__ float s_bh[32];
  __shared__ float s_eff[WPB][2][16][32];

  int tid = threadIdx.x;
  for (int i = tid; i < 320; i += 256) s_wrh[i] = Wr_h[i];
  if (tid < 32) s_bh[tid] = br_h[tid];
  __syncthreads();

  int wv = tid >> 5, lane = tid & 31;
  long ntiles = (long)((E + 15) >> 4);
  long tile = (long)blockIdx.x * WPB + wv;
  bool tlive = tile < ntiles;
  long tt = tlive ? tile : (ntiles - 1);
  int e = (int)(tt * 16) + (lane & 15);
  bool live = tlive && (e < E);
  int ec = (e < E) ? e : (E - 1);
  int src = esrc[ec], dst = edst[ec];

  // ---- geometry + spherical harmonics (block-normalized, 'norm') ----
  float vx = pos[3 * dst + 0] - pos[3 * src + 0];
  float vy = pos[3 * dst + 1] - pos[3 * src + 1];
  float vz = pos[3 * dst + 2] - pos[3 * src + 2];
  float r = sqrtf(vx * vx + vy * vy + vz * vz);
  float ir = 1.0f / (r + 1e-9f);
  float x = vx * ir, y = vy * ir, z = vz * ir;
  float sh[16];
  sh[0] = 1.0f / (1.0f + 1e-9f);
  {
    float nn = sqrtf(x * x + y * y + z * z) + 1e-9f;
    float iv = 1.0f / nn;
    sh[1] = x * iv; sh[2] = y * iv; sh[3] = z * iv;
  }
  {
    const float s3 = 1.7320508075688772f;
    float a0 = s3 * x * y, a1 = s3 * y * z, a2 = 0.5f * (3.0f * z * z - 1.0f);
    float a3 = s3 * x * z, a4 = 0.5f * s3 * (x * x - y * y);
    float nn = sqrtf(a0 * a0 + a1 * a1 + a2 * a2 + a3 * a3 + a4 * a4) + 1e-9f;
    float iv = 1.0f / nn;
    sh[4] = a0 * iv; sh[5] = a1 * iv; sh[6] = a2 * iv; sh[7] = a3 * iv; sh[8] = a4 * iv;
  }
  {
    const float c1 = 1.0458250331675945f, c2 = 5.123475382979799f;
    const float c3 = 0.8100925873009825f, c4 = 0.6614378277661477f;
    const float c5 = 2.5617376914898995f;
    float b0 = c1 * y * (3.0f * x * x - y * y);
    float b1 = c2 * x * y * z;
    float b2 = c3 * y * (5.0f * z * z - 1.0f);
    float b3 = c4 * (5.0f * z * z * z - 3.0f * z);
    float b4 = c3 * x * (5.0f * z * z - 1.0f);
    float b5 = c5 * (x * x - y * y) * z;
    float b6 = c1 * x * (x * x - 3.0f * y * y);
    float nn = sqrtf(b0 * b0 + b1 * b1 + b2 * b2 + b3 * b3 + b4 * b4 + b5 * b5 + b6 * b6) + 1e-9f;
    float iv = 1.0f / nn;
    sh[9] = b0 * iv; sh[10] = b1 * iv; sh[11] = b2 * iv; sh[12] = b3 * iv;
    sh[13] = b4 * iv; sh[14] = b5 * iv; sh[15] = b6 * iv;
  }

  // ---- radial basis * cutoff, radial MLP h = silu(rbf @ Wr_h + b) ----
  float rb[10];
  {
    float tcl = r * 0.1f; tcl = tcl > 1.0f ? 1.0f : tcl;
    float cut = 0.5f * (__cosf(3.14159265358979323846f * tcl) + 1.0f);
#pragma unroll
    for (int i = 0; i < 10; ++i) {
      float d = r - (10.0f / 9.0f) * (float)i;  // sigma = 1
      rb[i] = __expf(-0.5f * d * d) * cut;
    }
  }
  float h[32];
#pragma unroll
  for (int j = 0; j < 32; ++j) {
    float a = s_bh[j];
#pragma unroll
    for (int i = 0; i < 10; ++i) a += rb[i] * s_wrh[i * 32 + j];
    h[j] = a / (1.0f + __expf(-a));  // silu
  }
  float fs[3] = {feat[3 * src], feat[3 * src + 1], feat[3 * src + 2]};

  // ---- build fp16 A fragments of g = h (x) fs (16 edges x 96, ISA 16-bit A layout).
  // Branch on lane half so every kk (and thus h[]/fs[] index) is a compile-time
  // constant -> no runtime div/mod or cndmask selection trees.
  v16h A[3];
  if (lane < 16) {
#pragma unroll
    for (int kt = 0; kt < 3; ++kt) {
#pragma unroll
      for (int i = 0; i < 16; ++i) {
        int kk = kt * 32 + ((i >= 8) ? 16 : 0) + (i & 7);      // literal after unroll
        A[kt][i] = (_Float16)(h[kk / 3] * fs[kk % 3]);
      }
    }
  } else {
#pragma unroll
    for (int kt = 0; kt < 3; ++kt) {
#pragma unroll
      for (int i = 0; i < 16; ++i) {
        int kk = kt * 32 + 8 + ((i >= 8) ? 16 : 0) + (i & 7);  // literal after unroll
        A[kt][i] = (_Float16)(h[kk / 3] * fs[kk % 3]);
      }
    }
  }

  // ---- WMMA: eff = g @ W' (k and, in FULL pass, v); dump D frags to LDS ----
  const v16h* F = (const v16h*)frag;
  const int MM = FULL ? 2 : 1, NNT = FULL ? 2 : 1;
  for (int m = 0; m < MM; ++m) {
    for (int nt = 0; nt < NNT; ++nt) {
      v8f C = {};
#pragma unroll
      for (int kt = 0; kt < 3; ++kt) {
        v16h B = F[(((m * 3 + kt) * 2 + nt) << 5) + lane];
        C = __builtin_amdgcn_wmma_f32_16x16x32_f16(false, A[kt], false, B,
                                                   (short)0, C, false, false);
      }
      int mb = (lane >= 16) ? 8 : 0, ch = (lane & 15) + nt * 16;
#pragma unroll
      for (int i = 0; i < 8; ++i) s_eff[wv][m][mb + i][ch] = C[i];
    }
  }
  __syncthreads();

  // ---- per-edge logit / softmax scatter ----
  const float INV_SQRT3 = 0.5773502691896258f;
  const float INV_SQRT10 = 0.31622776601683794f;
  int em = lane & 15;
  const float* effk = s_eff[wv][0][em];
  float dot = 0.0f;
#pragma unroll
  for (int u = 0; u < 10; ++u) dot += effk[u] * q[dst * 10 + u];
  float logit = dot * sh[0] * INV_SQRT3 * INV_SQRT10;

  if (!FULL) {
    if (live && lane < 16) atomicMax(mxk + dst, fkey(logit));
  } else {
    float mxv = fdec(mxk[dst]);
    float ex = __expf(logit - mxv);
    if (live) {
      int half = lane >> 4;  // lane pair splits the mul channels of each irrep
      if (lane < 16) atomicAdd(den + dst, ex);
      const float* effv = s_eff[wv][1][em];
      float* ap = acc + (size_t)dst * 86;
      float exs = ex * INV_SQRT3;
      // l=0: channels 0..9, sh[0]
      float w0 = exs * sh[0];
      for (int u = half; u < 10; u += 2)
        atomicAdd(ap + u, effv[u] * w0);
      // l=1: base 10, 10 muls x 3 comps, sh[1..3] (constant indices)
      for (int u = half; u < 10; u += 2) {
        float w = effv[10 + u] * exs;
        float* p = ap + 10 + u * 3;
        atomicAdd(p + 0, w * sh[1]);
        atomicAdd(p + 1, w * sh[2]);
        atomicAdd(p + 2, w * sh[3]);
      }
      // l=2: base 40, 5 muls x 5 comps, sh[4..8]
      for (int u = half; u < 5; u += 2) {
        float w = effv[20 + u] * exs;
        float* p = ap + 40 + u * 5;
        atomicAdd(p + 0, w * sh[4]);
        atomicAdd(p + 1, w * sh[5]);
        atomicAdd(p + 2, w * sh[6]);
        atomicAdd(p + 3, w * sh[7]);
        atomicAdd(p + 4, w * sh[8]);
      }
      // l=3: base 65, 3 muls x 7 comps, sh[9..15]
      for (int u = half; u < 3; u += 2) {
        float w = effv[25 + u] * exs;
        float* p = ap + 65 + u * 7;
        atomicAdd(p + 0, w * sh[9]);
        atomicAdd(p + 1, w * sh[10]);
        atomicAdd(p + 2, w * sh[11]);
        atomicAdd(p + 3, w * sh[12]);
        atomicAdd(p + 4, w * sh[13]);
        atomicAdd(p + 5, w * sh[14]);
        atomicAdd(p + 6, w * sh[15]);
      }
    }
  }
}

// ---------------- finalize: normalize softmax, self-interaction, eq layernorm ----------------
__global__ void kFinal(const float* __restrict__ acc, const float* __restrict__ den,
                       const float* __restrict__ feat, const float* __restrict__ Wsi,
                       float* __restrict__ out, int N) {
  int n = blockIdx.x * blockDim.x + threadIdx.x;
  if (n >= N) return;
  float o[86];
  float iv = 1.0f / (den[n] + 1e-9f);
#pragma unroll
  for (int i = 0; i < 86; ++i) o[i] = acc[(size_t)n * 86 + i] * iv;
  float f0 = feat[3 * n], f1 = feat[3 * n + 1], f2 = feat[3 * n + 2];
#pragma unroll
  for (int u = 0; u < 10; ++u) o[u] += f0 * Wsi[u] + f1 * Wsi[10 + u] + f2 * Wsi[20 + u];
  // l=0: subtract mean over muls, rms-normalize
  float mean = 0.0f;
#pragma unroll
  for (int u = 0; u < 10; ++u) mean += o[u];
  mean *= 0.1f;
  float n2 = 0.0f;
#pragma unroll
  for (int u = 0; u < 10; ++u) { o[u] -= mean; n2 += o[u] * o[u]; }
  n2 *= 0.1f;
  float s = 1.0f / (sqrtf(n2) + 1e-6f);
#pragma unroll
  for (int u = 0; u < 10; ++u) o[u] *= s;
  // l=1: 30 vals, mean over 10 muls
  n2 = 0.0f;
#pragma unroll
  for (int i = 10; i < 40; ++i) n2 += o[i] * o[i];
  n2 *= 0.1f; s = 1.0f / (sqrtf(n2) + 1e-6f);
#pragma unroll
  for (int i = 10; i < 40; ++i) o[i] *= s;
  // l=2: 25 vals, mean over 5 muls
  n2 = 0.0f;
#pragma unroll
  for (int i = 40; i < 65; ++i) n2 += o[i] * o[i];
  n2 *= 0.2f; s = 1.0f / (sqrtf(n2) + 1e-6f);
#pragma unroll
  for (int i = 40; i < 65; ++i) o[i] *= s;
  // l=3: 21 vals, mean over 3 muls
  n2 = 0.0f;
#pragma unroll
  for (int i = 65; i < 86; ++i) n2 += o[i] * o[i];
  n2 *= (1.0f / 3.0f); s = 1.0f / (sqrtf(n2) + 1e-6f);
#pragma unroll
  for (int i = 65; i < 86; ++i) o[i] *= s;
#pragma unroll
  for (int i = 0; i < 86; ++i) out[(size_t)n * 86 + i] = o[i];
}

extern "C" void kernel_launch(void* const* d_in, const int* in_sizes, int n_in,
                              void* d_out, int out_size, void* d_ws, size_t ws_size,
                              hipStream_t stream) {
  const float* pos  = (const float*)d_in[0];
  const float* feat = (const float*)d_in[1];
  const float* Wq   = (const float*)d_in[2];
  const float* Wsi  = (const float*)d_in[3];
  const float* Wr_h = (const float*)d_in[4];
  const float* br_h = (const float*)d_in[5];
  const float* Wr_k = (const float*)d_in[6];
  const float* Wr_v = (const float*)d_in[7];
  const int*   esrc = (const int*)d_in[8];
  const int*   edst = (const int*)d_in[9];
  int N = in_sizes[0] / 3;
  int E = in_sizes[8];

  char* ws = (char*)d_ws;
  size_t off = 0;
  auto take = [&](size_t bytes) -> char* {
    char* p = ws + off;
    off += (bytes + 255) & ~(size_t)255;
    return p;
  };
  float*    acc  = (float*)take((size_t)N * 86 * sizeof(float));
  float*    den  = (float*)take((size_t)N * sizeof(float));
  int*      mxk  = (int*)take((size_t)N * sizeof(int));
  float*    q    = (float*)take((size_t)N * 10 * sizeof(float));
  _Float16* frag = (_Float16*)take((size_t)6144 * sizeof(_Float16));

  int initBlocks = (N * 86 + 255) / 256;
  kInit<<<initBlocks, 256, 0, stream>>>(acc, den, mxk, q, feat, Wq, N);
  kFrags<<<24, 256, 0, stream>>>(frag, Wr_k, Wr_v);

  long ntiles = (long)((E + 15) >> 4);
  int eBlocks = (int)((ntiles + WPB - 1) / WPB);
  kEdge<false><<<eBlocks, 256, 0, stream>>>(pos, feat, Wr_h, br_h, esrc, edst, q,
                                            frag, acc, den, mxk, N, E);
  kEdge<true><<<eBlocks, 256, 0, stream>>>(pos, feat, Wr_h, br_h, esrc, edst, q,
                                           frag, acc, den, mxk, N, E);
  kFinal<<<(N + 127) / 128, 128, 0, stream>>>(acc, den, feat, Wsi, (float*)d_out, N);
}
